// MultiHeadAttention2_54726473285660
// MI455X (gfx1250) — matrix operand order repaired and checked by reading, not compiled, not run on hardware
//
#include <hip/hip_runtime.h>

// Problem constants (from reference)
#define N_B   4
#define T_S   2048
#define D_M   1024
#define H_N   16
#define DH    64
#define NT    (N_B * T_S)          // 8192 rows in the flattened projection GEMMs
#define SCALE 0.03125f             // 1 / sqrt(1024)

typedef __bf16 bf16_t;
typedef bf16_t v16bf __attribute__((ext_vector_type(16)));
typedef float  v8f   __attribute__((ext_vector_type(8)));

union Frag {
  v16bf v;
  uint4 q[2];
  unsigned short us[16];
};

__device__ __forceinline__ unsigned short f2bf(float f) {
  union { float f; unsigned int u; } x;
  x.f = f;
  unsigned int u = x.u;
  // round-to-nearest-even f32 -> bf16
  return (unsigned short)((u + 0x7FFFu + ((u >> 16) & 1u)) >> 16);
}

__device__ __forceinline__ v8f wmma_bf16(const Frag& a, const Frag& b, v8f c) {
  return __builtin_amdgcn_wmma_f32_16x16x32_bf16(
      /*neg_a=*/false, a.v, /*neg_b=*/false, b.v,
      /*c_mod=*/(short)0, c, /*reuse_a=*/false, /*reuse_b=*/false);
}

// ---------------------------------------------------------------------------
// Kernel 1: projection GEMM  C[8192,1024] = A[8192,1024] @ W[1024,1024]
// blockIdx.z selects (query@Wq)->Qp, (keys@Wk)->Kp, (keys@Wv)->Vp.
//
// Block tile 128(M) x 128(N), 256 threads = 8 waves in a 4x2 grid; each wave
// computes 32x64 (8 WMMA accumulators). K-step 32. Double-buffered LDS
// (ping-pong): the next K-slab's global float4s are staged in registers before
// the current slab's WMMAs, then converted/stored to the other LDS buffer, so
// HBM latency hides under the WMMA stream with ONE barrier per K-step.
// All fragments load as 16B-aligned ds_load_b128 (LDS rows padded to 40 elems).
// ---------------------------------------------------------------------------
__global__ __launch_bounds__(256)
void mha_proj_gemm(const float* __restrict__ query,
                   const float* __restrict__ keys,
                   const float* __restrict__ Wq,
                   const float* __restrict__ Wk,
                   const float* __restrict__ Wv,
                   unsigned short* __restrict__ Qp,
                   unsigned short* __restrict__ Kp,
                   unsigned short* __restrict__ Vp) {
  const int z = blockIdx.z;
  const float* A = (z == 0) ? query : keys;
  const float* W = (z == 0) ? Wq : (z == 1) ? Wk : Wv;
  unsigned short* C = (z == 0) ? Qp : (z == 1) ? Kp : Vp;

  const int rowBase = blockIdx.x * 128;
  const int colBase = blockIdx.y * 128;
  const int tid  = threadIdx.x;
  const int wave = tid >> 5;
  const int lane = tid & 31;
  const int ml   = lane & 15;   // row within 16 (A layout) / col (B,C layout)
  const int kh   = lane >> 4;   // lane-half selector
  const int m0   = (wave >> 1) * 32;   // wave's M offset within block tile
  const int n0   = (wave & 1) * 64;    // wave's N offset within block tile

  // Ping-pong LDS, rows padded to 40 elems (80B) for 16B-aligned b128 frags
  __shared__ __align__(16) unsigned short As[2][128 * 40];  // [row][k]
  __shared__ __align__(16) unsigned short Bs[2][128 * 40];  // [n][k] (transposed)

  v8f acc[2][4];
  #pragma unroll
  for (int mi = 0; mi < 2; ++mi)
    #pragma unroll
    for (int t = 0; t < 4; ++t)
      #pragma unroll
      for (int e = 0; e < 8; ++e) acc[mi][t][e] = 0.0f;

  // Per-thread load coordinates (constant across K-steps)
  // A tile 128x32 = 1024 float4: idx = tid + j*256 -> r = idx>>3, c4 = idx&7
  // W tile 32x128 = 1024 float4: idx = tid + j*256 -> kr = idx>>5, c4 = idx&31
  // ---- prologue: tile 0 straight into buffer 0 ----
  {
    #pragma unroll
    for (int j = 0; j < 4; ++j) {
      int idx = tid + j * 256;
      int r = idx >> 3, c4 = idx & 7;
      float4 v = *(const float4*)(A + (size_t)(rowBase + r) * D_M + c4 * 4);
      ushort4 hb;
      hb.x = f2bf(v.x); hb.y = f2bf(v.y); hb.z = f2bf(v.z); hb.w = f2bf(v.w);
      *(ushort4*)(&As[0][r * 40 + c4 * 4]) = hb;
    }
    #pragma unroll
    for (int j = 0; j < 4; ++j) {
      int idx = tid + j * 256;
      int kr = idx >> 5, c4 = idx & 31;
      float4 v = *(const float4*)(W + (size_t)kr * D_M + colBase + c4 * 4);
      Bs[0][(c4 * 4 + 0) * 40 + kr] = f2bf(v.x);
      Bs[0][(c4 * 4 + 1) * 40 + kr] = f2bf(v.y);
      Bs[0][(c4 * 4 + 2) * 40 + kr] = f2bf(v.z);
      Bs[0][(c4 * 4 + 3) * 40 + kr] = f2bf(v.w);
    }
  }
  __syncthreads();

  for (int kt = 0; kt < D_M / 32; ++kt) {
    const int cur = kt & 1;
    const int nxt = cur ^ 1;
    const bool hasNext = (kt + 1) < (D_M / 32);

    // ---- stage next K-slab's global data in registers (overlaps WMMAs) ----
    float4 aReg[4], bReg[4];
    if (hasNext) {
      const int k1 = (kt + 1) * 32;
      #pragma unroll
      for (int j = 0; j < 4; ++j) {
        int idx = tid + j * 256;
        int r = idx >> 3, c4 = idx & 7;
        aReg[j] = *(const float4*)(A + (size_t)(rowBase + r) * D_M + k1 + c4 * 4);
      }
      #pragma unroll
      for (int j = 0; j < 4; ++j) {
        int idx = tid + j * 256;
        int kr = idx >> 5, c4 = idx & 31;
        bReg[j] = *(const float4*)(W + (size_t)(k1 + kr) * D_M + colBase + c4 * 4);
      }
    }

    // ---- compute on current buffer: batch all frag loads, then 8 WMMAs ----
    Frag af[2], bfr[4];
    #pragma unroll
    for (int mi = 0; mi < 2; ++mi) {
      const int arow = (m0 + mi * 16 + ml) * 40;
      af[mi].q[0] = *(const uint4*)(&As[cur][arow + 16 * kh]);
      af[mi].q[1] = *(const uint4*)(&As[cur][arow + 16 * kh + 8]);
    }
    #pragma unroll
    for (int t = 0; t < 4; ++t) {
      const int brow = (n0 + t * 16 + ml) * 40;
      bfr[t].q[0] = *(const uint4*)(&Bs[cur][brow + 16 * kh]);
      bfr[t].q[1] = *(const uint4*)(&Bs[cur][brow + 16 * kh + 8]);
    }
    #pragma unroll
    for (int mi = 0; mi < 2; ++mi)
      #pragma unroll
      for (int t = 0; t < 4; ++t)
        acc[mi][t] = wmma_bf16(af[mi], bfr[t], acc[mi][t]);

    // ---- convert staged registers into the other LDS buffer ----
    if (hasNext) {
      #pragma unroll
      for (int j = 0; j < 4; ++j) {
        int idx = tid + j * 256;
        int r = idx >> 3, c4 = idx & 7;
        ushort4 hb;
        hb.x = f2bf(aReg[j].x); hb.y = f2bf(aReg[j].y);
        hb.z = f2bf(aReg[j].z); hb.w = f2bf(aReg[j].w);
        *(ushort4*)(&As[nxt][r * 40 + c4 * 4]) = hb;
      }
      #pragma unroll
      for (int j = 0; j < 4; ++j) {
        int idx = tid + j * 256;
        int kr = idx >> 5, c4 = idx & 31;
        Bs[nxt][(c4 * 4 + 0) * 40 + kr] = f2bf(bReg[j].x);
        Bs[nxt][(c4 * 4 + 1) * 40 + kr] = f2bf(bReg[j].y);
        Bs[nxt][(c4 * 4 + 2) * 40 + kr] = f2bf(bReg[j].z);
        Bs[nxt][(c4 * 4 + 3) * 40 + kr] = f2bf(bReg[j].w);
      }
    }
    __syncthreads();   // next buffer ready; current buffer's reads retired
  }

  // C/D layout: row = i + 8*kh, col = t*16 + ml
  #pragma unroll
  for (int mi = 0; mi < 2; ++mi)
    #pragma unroll
    for (int t = 0; t < 4; ++t)
      #pragma unroll
      for (int i = 0; i < 8; ++i) {
        int row = rowBase + m0 + mi * 16 + i + 8 * kh;
        int col = colBase + n0 + t * 16 + ml;
        C[(size_t)row * D_M + col] = f2bf(acc[mi][t][i]);
      }
}

// ---------------------------------------------------------------------------
// Kernel 2: flash attention. One block per (n, h, 64-query tile); 128 threads
// = 4 waves; each wave owns 16 query rows. Iterates 64-key KV tiles with an
// online softmax; all matmuls are bf16 WMMA, stats/accumulators in f32.
// Next KV tile is prefetched (global_prefetch_b8) during compute.
// ---------------------------------------------------------------------------
__global__ __launch_bounds__(128)
void mha_flash_attn(const unsigned short* __restrict__ Qp,
                    const unsigned short* __restrict__ Kp,
                    const unsigned short* __restrict__ Vp,
                    float* __restrict__ out) {
  const int qt = blockIdx.x;    // query tile   0..31
  const int h  = blockIdx.y;    // head         0..15
  const int n  = blockIdx.z;    // batch        0..3
  const int tid  = threadIdx.x;
  const int wave = tid >> 5;
  const int lane = tid & 31;
  const int nl = lane & 15;
  const int kh = lane >> 4;

  __shared__ __align__(16) unsigned short Ks[64 * 72];       // [key][feat]
  __shared__ __align__(16) unsigned short Vs[64 * 72];       // [feat][key] (transposed)
  __shared__ __align__(16) unsigned short Ps[4 * 16 * 72];   // per-wave P staging

  // Preload Q fragments (A layout) straight from global: row = lane&15
  const int qrow = qt * 64 + wave * 16 + nl;
  const size_t qbase = ((size_t)(n * T_S) + qrow) * D_M + h * DH;
  Frag qf[2];
  #pragma unroll
  for (int kk = 0; kk < 2; ++kk) {
    qf[kk].q[0] = *(const uint4*)(Qp + qbase + kk * 32 + 16 * kh);
    qf[kk].q[1] = *(const uint4*)(Qp + qbase + kk * 32 + 16 * kh + 8);
  }

  float mrow[8], lrow[8];
  v8f o[4];
  #pragma unroll
  for (int i = 0; i < 8; ++i) { mrow[i] = -3.0e38f; lrow[i] = 0.0f; }
  #pragma unroll
  for (int t = 0; t < 4; ++t)
    #pragma unroll
    for (int e = 0; e < 8; ++e) o[t][e] = 0.0f;

  unsigned short* pw = &Ps[wave * 16 * 72];

  for (int kv = 0; kv < T_S / 64; ++kv) {
    __syncthreads();
    const size_t kvbase = ((size_t)(n * T_S) + kv * 64) * D_M + h * DH;
    // K tile key-major; V tile transposed to feature-major (512 uint4 / 128 thr)
    #pragma unroll
    for (int j = 0; j < 4; ++j) {
      int idx = tid + j * 128;
      int r = idx >> 3;               // key 0..63
      int c = idx & 7;                // 8-bf16 chunk 0..7
      uint4 kvec = *(const uint4*)(Kp + kvbase + (size_t)r * D_M + c * 8);
      *(uint4*)(&Ks[r * 72 + c * 8]) = kvec;
      union { uint4 q; unsigned short us[8]; } vu;
      vu.q = *(const uint4*)(Vp + kvbase + (size_t)r * D_M + c * 8);
      #pragma unroll
      for (int e = 0; e < 8; ++e)
        Vs[(c * 8 + e) * 72 + r] = vu.us[e];
    }
    __syncthreads();

    // Prefetch next KV tile into cache while this tile computes
    if (kv + 1 < T_S / 64) {
      const size_t nb = kvbase + (size_t)64 * D_M;
      const size_t poff = (size_t)(tid >> 1) * D_M + (tid & 1) * 32;
      __builtin_prefetch(Kp + nb + poff, 0, 3);
      __builtin_prefetch(Vp + nb + poff, 0, 3);
    }

    // S = Q @ K^T  (B matrix = K^T: n = key, K-dim = feature -> contiguous)
    v8f s[4];
    #pragma unroll
    for (int t = 0; t < 4; ++t)
      #pragma unroll
      for (int e = 0; e < 8; ++e) s[t][e] = 0.0f;
    #pragma unroll
    for (int kk = 0; kk < 2; ++kk) {
      Frag bfr[4];
      #pragma unroll
      for (int t = 0; t < 4; ++t) {
        const int base = (t * 16 + nl) * 72 + kk * 32 + 16 * kh;
        bfr[t].q[0] = *(const uint4*)(&Ks[base]);
        bfr[t].q[1] = *(const uint4*)(&Ks[base + 8]);
      }
      #pragma unroll
      for (int t = 0; t < 4; ++t)
        s[t] = wmma_bf16(qf[kk], bfr[t], s[t]);
    }

    // Online softmax. Row r = i + 8*kh lives on the 16 lanes sharing kh;
    // xor masks 1,2,4,8 butterfly-reduce exactly that group (wave32).
    #pragma unroll
    for (int i = 0; i < 8; ++i) {
      float v0 = s[0][i] * SCALE, v1 = s[1][i] * SCALE;
      float v2 = s[2][i] * SCALE, v3 = s[3][i] * SCALE;
      float rm = fmaxf(fmaxf(v0, v1), fmaxf(v2, v3));
      #pragma unroll
      for (int off = 1; off < 16; off <<= 1)
        rm = fmaxf(rm, __shfl_xor(rm, off, 32));
      float mn = fmaxf(mrow[i], rm);
      float alpha = __expf(mrow[i] - mn);
      mrow[i] = mn;
      float p0 = __expf(v0 - mn), p1 = __expf(v1 - mn);
      float p2 = __expf(v2 - mn), p3 = __expf(v3 - mn);
      float rs = (p0 + p1) + (p2 + p3);
      #pragma unroll
      for (int off = 1; off < 16; off <<= 1)
        rs += __shfl_xor(rs, off, 32);
      lrow[i] = lrow[i] * alpha + rs;
      s[0][i] = p0; s[1][i] = p1; s[2][i] = p2; s[3][i] = p3;
      #pragma unroll
      for (int t = 0; t < 4; ++t) o[t][i] *= alpha;
    }

    // Stage P (C layout) -> per-wave LDS strip, re-read in A layout.
    #pragma unroll
    for (int t = 0; t < 4; ++t)
      #pragma unroll
      for (int i = 0; i < 8; ++i)
        pw[(i + 8 * kh) * 72 + t * 16 + nl] = f2bf(s[t][i]);

    // Wave-private LDS round trip: split-counter wait, no workgroup barrier.
    asm volatile("s_wait_dscnt 0" ::: "memory");

    // O += P @ V  (B matrix = V: n = feature, K-dim = key -> contiguous in Vs)
    #pragma unroll
    for (int kk = 0; kk < 2; ++kk) {
      Frag pf;
      pf.q[0] = *(const uint4*)(&pw[nl * 72 + kk * 32 + 16 * kh]);
      pf.q[1] = *(const uint4*)(&pw[nl * 72 + kk * 32 + 16 * kh + 8]);
      Frag vf[4];
      #pragma unroll
      for (int t = 0; t < 4; ++t) {
        const int base = (t * 16 + nl) * 72 + kk * 32 + 16 * kh;
        vf[t].q[0] = *(const uint4*)(&Vs[base]);
        vf[t].q[1] = *(const uint4*)(&Vs[base + 8]);
      }
      #pragma unroll
      for (int t = 0; t < 4; ++t)
        o[t] = wmma_bf16(pf, vf[t], o[t]);
    }
  }

  // Epilogue: normalize and write f32 output [N,T,D]
  #pragma unroll
  for (int t = 0; t < 4; ++t)
    #pragma unroll
    for (int i = 0; i < 8; ++i) {
      int q   = qt * 64 + wave * 16 + i + 8 * kh;
      int col = h * DH + t * 16 + nl;
      out[((size_t)(n * T_S) + q) * D_M + col] = o[t][i] / lrow[i];
    }
}

// ---------------------------------------------------------------------------
extern "C" void kernel_launch(void* const* d_in, const int* in_sizes, int n_in,
                              void* d_out, int out_size, void* d_ws, size_t ws_size,
                              hipStream_t stream) {
  (void)in_sizes; (void)n_in; (void)out_size; (void)ws_size;
  const float* query = (const float*)d_in[0];
  const float* keys  = (const float*)d_in[1];
  const float* Wq    = (const float*)d_in[2];
  const float* Wk    = (const float*)d_in[3];
  const float* Wv    = (const float*)d_in[4];
  float* out = (float*)d_out;

  // Workspace: projected Q/K/V in bf16, 16 MB each (48 MB total)
  unsigned short* Qp = (unsigned short*)d_ws;
  unsigned short* Kp = Qp + (size_t)NT * D_M;
  unsigned short* Vp = Kp + (size_t)NT * D_M;

  // 3 GEMMs fused into one launch via blockIdx.z
  mha_proj_gemm<<<dim3(NT / 128, D_M / 128, 3), 256, 0, stream>>>(
      query, keys, Wq, Wk, Wv, Qp, Kp, Vp);

  // Flash attention: (query tiles, heads, batch)
  mha_flash_attn<<<dim3(T_S / 64, H_N, N_B), 128, 0, stream>>>(Qp, Kp, Vp, out);
}
